// TransformerDecoder_76175539962231
// MI455X (gfx1250) — matrix operand order repaired and checked by reading, not compiled, not run on hardware
//
#include <hip/hip_runtime.h>
#include <hip/hip_bf16.h>
#include <math.h>

// ---------------- model constants ----------------
#define kB   32
#define kT   64
#define kD   512
#define kH   8
#define kHD  64
#define kL   4
#define kFF  2048
#define kV   128
#define kM   (kB * kT)     // 2048 tokens
#define k3D  (3 * kD)      // 1536

typedef __attribute__((ext_vector_type(16))) _Float16 v16h;
typedef __attribute__((ext_vector_type(8)))  float    v8f;

// activation codes
#define ACT_NONE 0
#define ACT_SILU 1
#define ACT_GELU 2

__device__ __forceinline__ float act_apply(float v, int act)
{
    if (act == ACT_SILU)      return v / (1.0f + __expf(-v));
    else if (act == ACT_GELU) return 0.5f * v * (1.0f + erff(v * 0.70710678f));
    return v;
}

// =====================================================================
// Weight pre-pack: W (NxK fp32, row-major) -> fragment-ordered f16.
// Packed index = ((nTile*kt + kTile)*32 + lane)*16 + e,
// holding W[nTile*16 + (lane&15)][kTile*32 + kpat(e, lane>>4)]  (B-fragment layout: lane=N, elem=K).
// Coalesced writes; runs once per launch.
// =====================================================================
__global__ void pack_w(const float* __restrict__ W, _Float16* __restrict__ Wp,
                       int N, int K)
{
    const int tid = blockIdx.x * 256 + threadIdx.x;
    if (tid >= N * K) return;
    const int kt = K >> 5;
    const int e  = tid & 15;
    const int l  = (tid >> 4) & 31;
    const int kT_ = (tid >> 9) % kt;
    const int nT_ = (tid >> 9) / kt;
    const int v = e >> 1, w = e & 1;
    const int kpat = ((v < 4) ? (v * 2) : (16 + (v - 4) * 2)) + (l >> 4) * 8 + w;
    Wp[tid] = (_Float16)W[(size_t)(nT_ * 16 + (l & 15)) * K + kT_ * 32 + kpat];
}

// =====================================================================
// Activation pre-pack: X (MxK fp32) -> A-fragment-ordered f16.
// Packed index = ((mTile*kt + kTile)*32 + lane)*16 + e, holding
// X[mTile*16 + (lane&15)][kTile*32 + kpat(e, lane>>4)]  (ISA 16-bit A layout).
// =====================================================================
__global__ void pack_a(const float* __restrict__ X, _Float16* __restrict__ Xp,
                       int M, int K)
{
    const int tid = blockIdx.x * 256 + threadIdx.x;
    if (tid >= M * K) return;
    const int kt = K >> 5;
    const int e  = tid & 15;
    const int l  = (tid >> 4) & 31;
    const int kT_ = (tid >> 9) % kt;
    const int mT_ = (tid >> 9) / kt;
    const int v = e >> 1, w = e & 1;
    const int kpat = ((v < 4) ? (v * 2) : (16 + (v - 4) * 2)) + (l >> 4) * 8 + w;
    Xp[tid] = (_Float16)X[(size_t)(mT_ * 16 + (l & 15)) * K + kT_ * 32 + kpat];
}

// =====================================================================
// Packed GEMM: Y = act(A @ W^T + bias), both operands pre-packed f16.
// One wave owns a 32x32 output tile (4 accumulators).  Inner loop:
// 4 contiguous v16h loads + 4 independent v_wmma_f32_16x16x32_f16.
// grid = (M/32, N/128), block = 128 (4 waves).  EXEC all ones.
// =====================================================================
__global__ void gemm_packed(const _Float16* __restrict__ Ap,
                            const _Float16* __restrict__ Wp,
                            const float* __restrict__ bias,
                            float* __restrict__ Y,
                            int N, int K, int act)
{
    const int wave = threadIdx.x >> 5;
    const int lane = threadIdx.x & 31;
    const int m0   = blockIdx.x * 32;
    const int n0   = blockIdx.y * 128 + wave * 32;
    const int mrow = lane & 15;
    const int hi   = lane >> 4;
    const int kt   = K >> 5;

    // fragment streams: tile stride along K is 32 lanes * 16 halfs = 512
    const _Float16* a0p = Ap + ((size_t)((m0 >> 4)     * kt) * 32 + lane) * 16;
    const _Float16* a1p = Ap + ((size_t)(((m0 >> 4)+1) * kt) * 32 + lane) * 16;
    const _Float16* b0p = Wp + ((size_t)((n0 >> 4)     * kt) * 32 + lane) * 16;
    const _Float16* b1p = Wp + ((size_t)(((n0 >> 4)+1) * kt) * 32 + lane) * 16;

    v8f acc00 = {}, acc01 = {}, acc10 = {}, acc11 = {};

    for (int kTi = 0; kTi < kt; ++kTi) {
        const v16h a0 = *(const v16h*)a0p;  a0p += 512;
        const v16h a1 = *(const v16h*)a1p;  a1p += 512;
        const v16h b0 = *(const v16h*)b0p;  b0p += 512;
        const v16h b1 = *(const v16h*)b1p;  b1p += 512;

        acc00 = __builtin_amdgcn_wmma_f32_16x16x32_f16(false, a0, false, b0, (short)0, acc00, false, false);
        acc01 = __builtin_amdgcn_wmma_f32_16x16x32_f16(false, a0, false, b1, (short)0, acc01, false, false);
        acc10 = __builtin_amdgcn_wmma_f32_16x16x32_f16(false, a1, false, b0, (short)0, acc10, false, false);
        acc11 = __builtin_amdgcn_wmma_f32_16x16x32_f16(false, a1, false, b1, (short)0, acc11, false, false);
    }

    // C/D layout: lane -> N = base + (lane&15); VGPR r -> M = base + r + 8*hi
    const int nA = n0 + mrow;
    const int nB = n0 + 16 + mrow;
    const float bnA = bias ? bias[nA] : 0.0f;
    const float bnB = bias ? bias[nB] : 0.0f;

#pragma unroll
    for (int r = 0; r < 8; ++r) {
        const int mA = m0 + r + 8 * hi;
        const int mB = m0 + 16 + r + 8 * hi;
        Y[(size_t)mA * N + nA] = act_apply(acc00[r] + bnA, act);
        Y[(size_t)mA * N + nB] = act_apply(acc01[r] + bnB, act);
        Y[(size_t)mB * N + nA] = act_apply(acc10[r] + bnA, act);
        Y[(size_t)mB * N + nB] = act_apply(acc11[r] + bnB, act);
    }
}

// =====================================================================
// Fallback GEMM (used only if workspace too small for packing):
// direct fp32 gather + on-the-fly cvt, 32x32 wave tile.
// =====================================================================
__global__ void gemm_xwt(const float* __restrict__ X,
                         const float* __restrict__ W,
                         const float* __restrict__ bias,
                         float* __restrict__ Y,
                         int M, int N, int K, int act)
{
    const int wave = threadIdx.x >> 5;
    const int lane = threadIdx.x & 31;
    const int m0   = blockIdx.x * 32;
    const int n0   = blockIdx.y * 128 + wave * 32;
    const int mrow = lane & 15;
    const int hi   = lane >> 4;

    v8f acc00 = {}, acc01 = {}, acc10 = {}, acc11 = {};

    const float* xrow0 = X + (size_t)(m0 + mrow) * K;
    const float* xrow1 = X + (size_t)(m0 + 16 + mrow) * K;

    for (int k0 = 0; k0 < K; k0 += 32) {
        v16h a0, a1;
#pragma unroll
        for (int e = 0; e < 16; ++e) {
            int v = e >> 1, w = e & 1;
            int kk = k0 + ((v < 4) ? (v * 2) : (16 + (v - 4) * 2)) + hi * 8 + w;
            a0[e] = (_Float16)xrow0[kk];
            a1[e] = (_Float16)xrow1[kk];
        }
        v16h b0, b1;
#pragma unroll
        for (int e = 0; e < 16; ++e) {
            int v = e >> 1, w = e & 1;
            int kk = k0 + ((v < 4) ? (v * 2) : (16 + (v - 4) * 2)) + hi * 8 + w;
            b0[e] = (_Float16)W[(size_t)(n0 + mrow) * K + kk];
            b1[e] = (_Float16)W[(size_t)(n0 + 16 + mrow) * K + kk];
        }
        acc00 = __builtin_amdgcn_wmma_f32_16x16x32_f16(false, a0, false, b0, (short)0, acc00, false, false);
        acc01 = __builtin_amdgcn_wmma_f32_16x16x32_f16(false, a0, false, b1, (short)0, acc01, false, false);
        acc10 = __builtin_amdgcn_wmma_f32_16x16x32_f16(false, a1, false, b0, (short)0, acc10, false, false);
        acc11 = __builtin_amdgcn_wmma_f32_16x16x32_f16(false, a1, false, b1, (short)0, acc11, false, false);
    }

    const int nA = n0 + mrow;
    const int nB = n0 + 16 + mrow;
    const float bnA = bias ? bias[nA] : 0.0f;
    const float bnB = bias ? bias[nB] : 0.0f;
#pragma unroll
    for (int r = 0; r < 8; ++r) {
        const int mA = m0 + r + 8 * hi;
        const int mB = m0 + 16 + r + 8 * hi;
        Y[(size_t)mA * N + nA] = act_apply(acc00[r] + bnA, act);
        Y[(size_t)mA * N + nB] = act_apply(acc01[r] + bnB, act);
        Y[(size_t)mB * N + nA] = act_apply(acc10[r] + bnA, act);
        Y[(size_t)mB * N + nB] = act_apply(acc11[r] + bnB, act);
    }
}

// =====================================================================
// Embedding + xPos rotary embedding.  grid = B*T rows, block = 256 (pairs).
// =====================================================================
__global__ void embed_xpos(const int* __restrict__ pred,
                           const float* __restrict__ emb,
                           float* __restrict__ x)
{
    const int row = blockIdx.x;            // b*T + t
    const int t   = row % kT;
    const int j   = threadIdx.x;           // pair index 0..255
    const int tok = pred[row];

    const float e0 = emb[(size_t)tok * kD + 2 * j];
    const float e1 = emb[(size_t)tok * kD + 2 * j + 1];

    const float sv   = (2.0f * j + 0.4f * kD) / (1.4f * kD);
    const float scl  = powf(sv, (float)t / 16.0f);
    const float invf = powf(10000.0f, -(float)j / 256.0f);
    const float ang  = (float)t * invf;
    const float sn = sinf(ang) * scl;
    const float cs = cosf(ang) * scl;

    x[(size_t)row * kD + 2 * j]     = e0 * cs - e1 * sn;
    x[(size_t)row * kD + 2 * j + 1] = e1 * cs + e0 * sn;
}

// =====================================================================
// Self-attention per (b,h): 64 q x 64 k x 64 d, causal + (pred==0) mask,
// two-pass softmax.  grid = B*H, block = 64.
// =====================================================================
__global__ void attention(const float* __restrict__ qkv,
                          const int*   __restrict__ pred,
                          float* __restrict__ o)
{
    const int bh = blockIdx.x;
    const int b  = bh / kH, h = bh % kH;
    const int tq = threadIdx.x;
    const float scale = 0.125f;

    const float* qrow = qkv + ((size_t)(b * kT + tq)) * k3D + h * kHD;

    float mx = -1e30f;
    for (int tk = 0; tk < kT; ++tk) {
        const float* krow = qkv + ((size_t)(b * kT + tk)) * k3D + kD + h * kHD;
        float d = 0.0f;
#pragma unroll
        for (int i = 0; i < kHD; ++i) d += qrow[i] * krow[i];
        d *= scale;
        if (tk > tq || pred[b * kT + tk] == 0) d = -1.0e9f;
        mx = fmaxf(mx, d);
    }

    float acc[kHD];
#pragma unroll
    for (int i = 0; i < kHD; ++i) acc[i] = 0.0f;
    float sum = 0.0f;

    for (int tk = 0; tk < kT; ++tk) {
        const float* krow = qkv + ((size_t)(b * kT + tk)) * k3D + kD + h * kHD;
        float d = 0.0f;
#pragma unroll
        for (int i = 0; i < kHD; ++i) d += qrow[i] * krow[i];
        d *= scale;
        if (tk > tq || pred[b * kT + tk] == 0) d = -1.0e9f;
        const float p = __expf(d - mx);
        sum += p;
        const float* vrow = qkv + ((size_t)(b * kT + tk)) * k3D + 2 * kD + h * kHD;
#pragma unroll
        for (int i = 0; i < kHD; ++i) acc[i] += p * vrow[i];
    }

    const float inv = 1.0f / sum;
    float* orow = o + ((size_t)(b * kT + tq)) * kD + h * kHD;
#pragma unroll
    for (int i = 0; i < kHD; ++i) orow[i] = acc[i] * inv;
}

// =====================================================================
// x = LayerNorm(x + resid)*g + b  (in place).  resid row = row/resid_div
// (div=T for the 1-token cross-attention broadcast); resid may be null.
// =====================================================================
__global__ void add_ln(float* __restrict__ x,
                       const float* __restrict__ resid, int resid_div,
                       const float* __restrict__ g,
                       const float* __restrict__ bb)
{
    const int row = blockIdx.x;
    const int tid = threadIdx.x;
    __shared__ float red[128];

    float vals[4];
    float s = 0.0f;
#pragma unroll
    for (int i = 0; i < 4; ++i) {
        const int d = tid + i * 128;
        float v = x[(size_t)row * kD + d];
        if (resid) v += resid[(size_t)(row / resid_div) * kD + d];
        vals[i] = v; s += v;
    }
    red[tid] = s; __syncthreads();
    for (int off = 64; off > 0; off >>= 1) {
        if (tid < off) red[tid] += red[tid + off];
        __syncthreads();
    }
    const float mu = red[0] / (float)kD;
    __syncthreads();

    float vs = 0.0f;
#pragma unroll
    for (int i = 0; i < 4; ++i) { const float d0 = vals[i] - mu; vs += d0 * d0; }
    red[tid] = vs; __syncthreads();
    for (int off = 64; off > 0; off >>= 1) {
        if (tid < off) red[tid] += red[tid + off];
        __syncthreads();
    }
    const float rstd = rsqrtf(red[0] / (float)kD + 1e-5f);

#pragma unroll
    for (int i = 0; i < 4; ++i) {
        const int d = tid + i * 128;
        x[(size_t)row * kD + d] = (vals[i] - mu) * rstd * g[d] + bb[d];
    }
}

// h[b,:] = x[b*T + (t-1), :]
__global__ void gather_h(const float* __restrict__ x, float* __restrict__ h, int t)
{
    const int b = blockIdx.x, d = threadIdx.x; // block = 512
    h[(size_t)b * kD + d] = x[((size_t)b * kT + (t - 1)) * kD + d];
}

__device__ __forceinline__ float u01(unsigned x)
{
    x ^= x >> 16; x *= 0x7feb352du;
    x ^= x >> 15; x *= 0x846ca68bu;
    x ^= x >> 16;
    return (float)(x >> 8) * (1.0f / 16777216.0f);
}

// Gumbel-max sampling.  grid = B, block = 128.
__global__ void sample(const float* __restrict__ logits, int* __restrict__ pred, int t)
{
    const int b = blockIdx.x, v = threadIdx.x;
    const float u = u01((unsigned)(t * 0x9E3779B9u) ^ (unsigned)(b * 0x85EBCA6Bu)
                        ^ (unsigned)(v * 0xC2B2AE35u) ^ 0x1234567u);
    const float g = -logf(-logf(u + 1e-9f) + 1e-9f);
    const float score = logits[b * kV + v] * 10.0f + g;   // /TEMP, TEMP=0.1

    __shared__ float sv_[128];
    __shared__ int   si_[128];
    sv_[v] = score; si_[v] = v; __syncthreads();
    for (int off = 64; off > 0; off >>= 1) {
        if (v < off && sv_[v + off] > sv_[v]) { sv_[v] = sv_[v + off]; si_[v] = si_[v + off]; }
        __syncthreads();
    }
    if (v == 0) pred[b * kT + t] = si_[0];
}

__global__ void init_pred(int* __restrict__ pred)
{
    const int i = blockIdx.x * 256 + threadIdx.x;
    if (i < kB * kT) pred[i] = (i % kT == 0) ? 1 : 0;
}

// =====================================================================
// Host-side orchestration
// =====================================================================
extern "C" void kernel_launch(void* const* d_in, const int* in_sizes, int n_in,
                              void* d_out, int out_size, void* d_ws, size_t ws_size,
                              hipStream_t stream)
{
    (void)in_sizes; (void)n_in; (void)out_size;

    const float* mol      = (const float*)d_in[1];
    const float* emb      = (const float*)d_in[2];
    const float* sa_in_w  = (const float*)d_in[3];
    const float* sa_in_b  = (const float*)d_in[4];
    const float* sa_out_w = (const float*)d_in[5];
    const float* sa_out_b = (const float*)d_in[6];
    const float* ca_in_w  = (const float*)d_in[7];
    const float* ca_in_b  = (const float*)d_in[8];
    const float* ca_out_w = (const float*)d_in[9];
    const float* ca_out_b = (const float*)d_in[10];
    const float* ff_w1    = (const float*)d_in[11];
    const float* ff_b1    = (const float*)d_in[12];
    const float* ff_w2    = (const float*)d_in[13];
    const float* ff_b2    = (const float*)d_in[14];
    const float* n1_g     = (const float*)d_in[15];
    const float* n1_b     = (const float*)d_in[16];
    const float* n2_g     = (const float*)d_in[17];
    const float* n2_b     = (const float*)d_in[18];
    const float* n3_g     = (const float*)d_in[19];
    const float* n3_b     = (const float*)d_in[20];
    const float* fn_g     = (const float*)d_in[21];
    const float* fn_b     = (const float*)d_in[22];
    const float* out_w1   = (const float*)d_in[23];
    const float* out_b1   = (const float*)d_in[24];
    const float* out_w2   = (const float*)d_in[25];
    const float* out_b2   = (const float*)d_in[26];
    const float* out_w3   = (const float*)d_in[27];
    const float* out_b3   = (const float*)d_in[28];

    int* pred = (int*)d_out;                 // (B,T) int32 — working token buffer

    // ---- fp32 activation carve ----
    float* p = (float*)d_ws;
    float* x      = p; p += (size_t)kM * kD;
    float* qkv    = p; p += (size_t)kM * k3D;
    float* attno  = p; p += (size_t)kM * kD;
    float* tmp    = p; p += (size_t)kM * kD;
    float* ff     = p; p += (size_t)kM * kFF;
    float* cv     = p; p += (size_t)kB * kD;
    float* cb     = p; p += (size_t)kL * kB * kD;   // per-layer cross-attn vector
    float* hb     = p; p += (size_t)kB * kD;
    float* h1     = p; p += (size_t)kB * 1024;
    float* h2     = p; p += (size_t)kB * 1024;
    float* logits = p; p += (size_t)kB * kV;

    const size_t actBytes = (size_t)((char*)p - (char*)d_ws);

    // ---- packed-f16 region ----
    const size_t szSaIn  = (size_t)k3D * kD;
    const size_t szDxD   = (size_t)kD * kD;
    const size_t szFf    = (size_t)kFF * kD;
    const size_t szOut1  = (size_t)1024 * kD;
    const size_t szOut2  = (size_t)1024 * 1024;
    const size_t szOut3  = (size_t)kV * 1024;
    const size_t szApack = (size_t)kM * kFF;        // max M*K among GEMMs

    const size_t packedHalfs = kL * (szSaIn + 3 * szDxD + 2 * szFf)
                             + szOut1 + szOut2 + szOut3 + szApack;
    const bool usePacked = ws_size >= actBytes + packedHalfs * sizeof(_Float16);

    _Float16* hp = (_Float16*)((char*)d_ws + actBytes);
    _Float16 *wpSaIn[kL], *wpSaOut[kL], *wpCaV[kL], *wpCaOut[kL], *wpFf1[kL], *wpFf2[kL];
    _Float16 *wpOut1 = nullptr, *wpOut2 = nullptr, *wpOut3 = nullptr, *apack = nullptr;
    if (usePacked) {
        for (int l = 0; l < kL; ++l) {
            wpSaIn[l]  = hp; hp += szSaIn;
            wpSaOut[l] = hp; hp += szDxD;
            wpCaV[l]   = hp; hp += szDxD;
            wpCaOut[l] = hp; hp += szDxD;
            wpFf1[l]   = hp; hp += szFf;
            wpFf2[l]   = hp; hp += szFf;
        }
        wpOut1 = hp; hp += szOut1;
        wpOut2 = hp; hp += szOut2;
        wpOut3 = hp; hp += szOut3;
        apack  = hp; hp += szApack;
    }

    auto packW = [&](const float* Wf, _Float16* Wp, int N, int K) {
        pack_w<<<(int)(((size_t)N * K + 255) / 256), 256, 0, stream>>>(Wf, Wp, N, K);
    };
    auto runGemm = [&](const float* X, const float* Wf, const _Float16* Wp,
                       const float* bias, float* Y, int M, int N, int K, int act) {
        dim3 g(M / 32, N / 128);
        if (usePacked) {
            pack_a<<<(int)(((size_t)M * K + 255) / 256), 256, 0, stream>>>(X, apack, M, K);
            gemm_packed<<<g, 128, 0, stream>>>(apack, Wp, bias, Y, N, K, act);
        } else {
            gemm_xwt<<<g, 128, 0, stream>>>(X, Wf, bias, Y, M, N, K, act);
        }
    };

    // ---- one-time weight packing (amortized over 63 steps) ----
    if (usePacked) {
        for (int l = 0; l < kL; ++l) {
            packW(sa_in_w  + (size_t)l * szSaIn,                         wpSaIn[l],  k3D, kD);
            packW(sa_out_w + (size_t)l * szDxD,                          wpSaOut[l], kD,  kD);
            packW(ca_in_w  + (size_t)l * szSaIn + (size_t)2 * kD * kD,   wpCaV[l],   kD,  kD); // wv slice
            packW(ca_out_w + (size_t)l * szDxD,                          wpCaOut[l], kD,  kD);
            packW(ff_w1    + (size_t)l * szFf,                           wpFf1[l],   kFF, kD);
            packW(ff_w2    + (size_t)l * szFf,                           wpFf2[l],   kD,  kFF);
        }
        packW(out_w1, wpOut1, 2 * kD, kD);
        packW(out_w2, wpOut2, 1024, 1024);
        packW(out_w3, wpOut3, kV, 1024);
    }

    init_pred<<<(kB * kT + 255) / 256, 256, 0, stream>>>(pred);

    // ---- cross-attention is step-invariant (1 memory token => softmax==1):
    //      precompute per-layer broadcast vector cb[l] once.
    for (int l = 0; l < kL; ++l) {
        runGemm(mol, ca_in_w + (size_t)l * szSaIn + (size_t)2 * kD * kD, wpCaV[l],
                ca_in_b + (size_t)l * k3D + 2 * kD, cv, kB, kD, kD, ACT_NONE);
        runGemm(cv, ca_out_w + (size_t)l * szDxD, wpCaOut[l],
                ca_out_b + (size_t)l * kD, cb + (size_t)l * kB * kD, kB, kD, kD, ACT_NONE);
    }

    for (int t = 1; t < kT; ++t) {
        embed_xpos<<<kM, 256, 0, stream>>>(pred, emb, x);

        for (int l = 0; l < kL; ++l) {
            runGemm(x, sa_in_w + (size_t)l * szSaIn, wpSaIn[l],
                    sa_in_b + (size_t)l * k3D, qkv, kM, k3D, kD, ACT_NONE);
            attention<<<kB * kH, kT, 0, stream>>>(qkv, pred, attno);
            runGemm(attno, sa_out_w + (size_t)l * szDxD, wpSaOut[l],
                    sa_out_b + (size_t)l * kD, tmp, kM, kD, kD, ACT_NONE);
            add_ln<<<kM, 128, 0, stream>>>(x, tmp, 1,
                                           n1_g + (size_t)l * kD, n1_b + (size_t)l * kD);

            add_ln<<<kM, 128, 0, stream>>>(x, cb + (size_t)l * kB * kD, kT,
                                           n2_g + (size_t)l * kD, n2_b + (size_t)l * kD);

            runGemm(x, ff_w1 + (size_t)l * szFf, wpFf1[l],
                    ff_b1 + (size_t)l * kFF, ff, kM, kFF, kD, ACT_SILU);
            runGemm(ff, ff_w2 + (size_t)l * szFf, wpFf2[l],
                    ff_b2 + (size_t)l * kD, tmp, kM, kD, kFF, ACT_NONE);
            add_ln<<<kM, 128, 0, stream>>>(x, tmp, 1,
                                           n3_g + (size_t)l * kD, n3_b + (size_t)l * kD);
        }

        add_ln<<<kM, 128, 0, stream>>>(x, nullptr, 1, fn_g, fn_b);  // final LN
        gather_h<<<kB, kD, 0, stream>>>(x, hb, t);

        runGemm(hb, out_w1, wpOut1, out_b1, h1, kB, 2 * kD, kD,   ACT_GELU);
        runGemm(h1, out_w2, wpOut2, out_b2, h2, kB, 1024,   1024, ACT_GELU);
        runGemm(h2, out_w3, wpOut3, out_b3, logits, kB, kV, 1024, ACT_NONE);

        sample<<<kB, kV, 0, stream>>>(logits, pred, t);
    }
}
